// SimulatedNetwork_45518063403458
// MI455X (gfx1250) — matrix operand order repaired and checked by reading, not compiled, not run on hardware
//
#include <hip/hip_runtime.h>

// Fused 8-layer MLP (D=64) for gfx1250 using v_wmma_f32_16x16x32_f16.
// Computes y.T = W * h.T per layer. Hidden neurons are relabeled by the
// involution sigma(k) = swap bits 3,4 of k, which is absorbed into the
// weight/x fetch addresses; this makes the D->B inter-layer relayout
// fully lane-local (no shuffles, no selects): just cvt + packed-f16 ReLU.

typedef _Float16 half_t;
typedef __attribute__((ext_vector_type(16))) _Float16     v16h;
typedef __attribute__((ext_vector_type(8)))  float        v8f;
typedef __attribute__((ext_vector_type(4)))  unsigned int v4u;

#define NLAYERS 8
#define DD 64

union HVec {
    v16h h;
    v4u  q[2];
    unsigned int u[8];
};

// Pack two f32 into one u32 holding two f16 (RNE converts -> v_cvt_f16_f32 .l/.h).
__device__ __forceinline__ unsigned int pack2(float a, float b) {
    _Float16 ha = (_Float16)a;
    _Float16 hb = (_Float16)b;
    unsigned short ua = __builtin_bit_cast(unsigned short, ha);
    unsigned short ub = __builtin_bit_cast(unsigned short, hb);
    return (unsigned int)ua | ((unsigned int)ub << 16);
}

// Packed f16 ReLU: single VOP3P v_pk_max_num_f16 against inline constant 0.
__device__ __forceinline__ unsigned int relu_pk(unsigned int u) {
    unsigned int r;
    asm("v_pk_max_num_f16 %0, %1, 0" : "=v"(r) : "v"(u));
    return r;
}

__global__ __launch_bounds__(256) void fused_mlp_kernel(
    const half_t* __restrict__ x,      // [B, 64] row-major
    const half_t* __restrict__ Wh,     // [7, 64, 64] row-major (y = h @ W.T)
    const half_t* __restrict__ Wlast,  // [64, 64]
    half_t* __restrict__ out)          // [B, 64]
{
    // All 8 layers of weights staged in LDS: 8*64*64 f16 = 64 KB (of 320 KB/WGP).
    __shared__ half_t wlds[NLAYERS * DD * DD];

    const int tid = threadIdx.x;

    // Cooperative weight staging (16-byte vectors).
    {
        v4u* dst = (v4u*)wlds;
        const v4u* s0 = (const v4u*)Wh;     // 7*64*64/8 = 3584 v4u
        const v4u* s1 = (const v4u*)Wlast;  //   64*64/8 =  512 v4u
        for (int i = tid; i < 3584; i += 256) dst[i] = s0[i];
        for (int i = tid; i < 512;  i += 256) dst[3584 + i] = s1[i];
    }
    __syncthreads();

    const int lane = tid & 31;
    const int wave = tid >> 5;
    const int l16  = lane & 15;   // B-matrix column (batch row in tile); A-matrix row
    const int lhi  = lane >> 4;   // which K half-group this lane holds

    // Each wave owns 64 batch rows (4 sub-tiles of 16).
    const int row0 = (blockIdx.x * 8 + wave) * 64;

    // Base pointers; bt/kt deltas fold into the 24-bit instruction offsets.
    // x is loaded pre-permuted by sigma: B element e of half-group lhi holds
    // feature kt*32 + (e>>3)*16 + lhi*8 + (e&7)  ->  chunks at +0 and +16.
    const half_t* xbase = x   + (size_t)(row0 + l16) * DD + lhi * 8;
    half_t*       obase = out + (size_t)(row0 + l16) * DD + lhi * 8;

    HVec Bm[4][2];  // [batch subtile][k-step]
#pragma unroll
    for (int bt = 0; bt < 4; ++bt) {
#pragma unroll
        for (int kt = 0; kt < 2; ++kt) {
            const half_t* p = xbase + bt * 16 * DD + kt * 32;
            Bm[bt][kt].q[0] = __builtin_nontemporal_load((const v4u*)p);
            Bm[bt][kt].q[1] = __builtin_nontemporal_load((const v4u*)(p + 16));
        }
    }

    v8f acc[4][4];  // [batch subtile][neuron m-tile], y^T tiles (M=neurons, N=batch)
    const v8f zero = {};

#pragma unroll
    for (int layer = 0; layer < NLAYERS; ++layer) {
        const half_t* wl = wlds + layer * DD * DD;

        // ---- GEMM: acc[bt][m] = W_tile(m,kt) x B(bt,kt), K = 64 in 2 steps ----
        // A element s of half lhi holds W[M, k] with k in table order:
        // row*64 + kt*32 + lhi*8 + 0..7, and row*64 + kt*32 + 16 + lhi*8 + 0..7.
#pragma unroll
        for (int m = 0; m < 4; ++m) {
            {
                const half_t* ap = wl + (m * 16 + l16) * DD + 0 * 32 + lhi * 8;
                HVec A;
                A.q[0] = *(const v4u*)ap;         // ds_load_b128
                A.q[1] = *(const v4u*)(ap + 16);  // ds_load_b128
#pragma unroll
                for (int bt = 0; bt < 4; ++bt)
                    acc[bt][m] = __builtin_amdgcn_wmma_f32_16x16x32_f16(
                        false, A.h, false, Bm[bt][0].h, (short)0, zero, false, false);
            }
            {
                const half_t* ap = wl + (m * 16 + l16) * DD + 1 * 32 + lhi * 8;
                HVec A;
                A.q[0] = *(const v4u*)ap;
                A.q[1] = *(const v4u*)(ap + 16);
#pragma unroll
                for (int bt = 0; bt < 4; ++bt)
                    acc[bt][m] = __builtin_amdgcn_wmma_f32_16x16x32_f16(
                        false, A.h, false, Bm[bt][1].h, (short)0, acc[bt][m], false, false);
            }
        }

        // ---- Relayout D (y^T, f32) -> next-layer B (f16): fully lane-local ----
        // Value at D(m, half, r) goes to Bm[m>>1], element (m&1)*8 + r, same lane.
        if (layer < NLAYERS - 1) {
#pragma unroll
            for (int bt = 0; bt < 4; ++bt) {
#pragma unroll
                for (int m = 0; m < 4; ++m) {
                    const int kt = m >> 1;
                    const int ub = (m & 1) * 4;
#pragma unroll
                    for (int j = 0; j < 4; ++j)
                        Bm[bt][kt].u[ub + j] =
                            relu_pk(pack2(acc[bt][m][2 * j], acc[bt][m][2 * j + 1]));
                }
            }
        }
    }

    // ---- Store final layer (no activation): out[b, n], b = N, n = M = p ----
#pragma unroll
    for (int bt = 0; bt < 4; ++bt) {
#pragma unroll
        for (int m = 0; m < 4; ++m) {
            v4u st;
#pragma unroll
            for (int j = 0; j < 4; ++j)
                st[j] = pack2(acc[bt][m][2 * j], acc[bt][m][2 * j + 1]);
            __builtin_nontemporal_store(st, (v4u*)(obase + bt * 16 * DD + m * 16));
        }
    }
}

extern "C" void kernel_launch(void* const* d_in, const int* in_sizes, int n_in,
                              void* d_out, int out_size, void* d_ws, size_t ws_size,
                              hipStream_t stream) {
    const half_t* x     = (const half_t*)d_in[0];  // [B, 64] fp16
    const half_t* Wh    = (const half_t*)d_in[1];  // [7, 64, 64] fp16
    const half_t* Wlast = (const half_t*)d_in[2];  // [64, 64] fp16
    half_t* out = (half_t*)d_out;

    const int rows = in_sizes[0] / DD;       // 1<<20
    const int rows_per_block = 8 * 64;       // 8 waves x 64 rows
    const int grid = rows / rows_per_block;  // 2048

    fused_mlp_kernel<<<grid, 256, 0, stream>>>(x, Wh, Wlast, out);
}
